// Block_83073257439390
// MI455X (gfx1250) — compile-verified, matrix operand order checked by reading, not run on hardware
//
#include <hip/hip_runtime.h>
#include <hip/hip_bf16.h>
#include <math.h>

// ---------------------------------------------------------------------------
// Transformer block for MI455X (gfx1250, wave32, WMMA bf16 16x16x32).
// B=64, T=512, C=512, H=16, D=32.
// GEMMs: 64x128 block tile, 32x64 per wave, double-buffered LDS fed by
// GLOBAL_LOAD_ASYNC_TO_LDS_B128 (ASYNCcnt), 8 WMMAs per wave per K-step.
// ---------------------------------------------------------------------------

typedef __bf16 bf16_t;
typedef __bf16 v16bf __attribute__((ext_vector_type(16)));
typedef __bf16 v8bf  __attribute__((ext_vector_type(8)));
typedef float  v8f   __attribute__((ext_vector_type(8)));

#define WMMA_BF16(A, B, C) \
  __builtin_amdgcn_wmma_f32_16x16x32_bf16(false, (A), false, (B), (short)0, (C), false, false)

constexpr int Tn = 512;
constexpr int Cn = 512;
constexpr int Hn = 16;
constexpr int Dn = 32;
constexpr int Bn = 64;
constexpr int Mrows = Bn * Tn;  // 32768

// ---------------------------------------------------------------------------
// Async global -> LDS copy (16 bytes per lane). VDST VGPR carries the LDS
// byte offset (low 32 bits of the generic shared pointer == LDS offset,
// since the shared aperture tag lives in addr[63:32]).  Tracked by ASYNCcnt.
// ---------------------------------------------------------------------------
__device__ inline void async_cp_b128(const bf16_t* gsrc, bf16_t* ldst) {
  unsigned loff = (unsigned)(size_t)ldst;
  asm volatile("global_load_async_to_lds_b128 %0, %1, off"
               :: "v"(loff), "v"(gsrc)
               : "memory");
}

__device__ inline void wait_async() {
  asm volatile("s_wait_asynccnt 0x0" ::: "memory");
}

// ---------------------------------------------------------------------------
// Fragment load from an LDS (or any) [rows][32] K-contiguous bf16 tile.
// Matches CDNA5 16-bit A/B operand layout for 16x16x32:
//   lane l: row = row0 + (l & 15)
//   halves j=0..7  -> K = 8*(l>>4) + j
//   halves j=8..15 -> K = 16 + 8*(l>>4) + (j-8)
// ---------------------------------------------------------------------------
__device__ inline v16bf ldfrag(const bf16_t* s, int row0, int lane) {
  const bf16_t* p = s + (size_t)(row0 + (lane & 15)) * 32 + ((lane >> 4) * 8);
  v8bf lo = *(const v8bf*)p;
  v8bf hi = *(const v8bf*)(p + 16);
  v16bf f;
#pragma unroll
  for (int i = 0; i < 8; ++i) { f[i] = lo[i]; f[i + 8] = hi[i]; }
  return f;
}

// Same fragment layout, loading straight from global with row stride ld.
__device__ inline v16bf ldfrag_g(const bf16_t* base, size_t ld, int row0, int lane) {
  const bf16_t* p = base + (size_t)(row0 + (lane & 15)) * ld + ((lane >> 4) * 8);
  v8bf lo = *(const v8bf*)p;
  v8bf hi = *(const v8bf*)(p + 16);
  v16bf f;
#pragma unroll
  for (int i = 0; i < 8; ++i) { f[i] = lo[i]; f[i + 8] = hi[i]; }
  return f;
}

// ---------------------------------------------------------------------------
// Weight convert fp32 -> bf16, stored transposed as (N, K) row-major so GEMM
// B-tiles are K-contiguous (identical LDS access pattern to A-tiles).
// HEADS=true handles (H, K, 32) layout (Wq/Wk/Wv) with n = h*32 + d.
// ---------------------------------------------------------------------------
template <bool HEADS>
__global__ __launch_bounds__(256) void cvtw_kernel(const float* __restrict__ W,
                                                   bf16_t* __restrict__ Wt,
                                                   int K, int N) {
  size_t idx = (size_t)blockIdx.x * 256 + threadIdx.x;
  int n = (int)(idx / K);
  int k = (int)(idx % K);
  float v = HEADS ? W[(size_t)(n >> 5) * K * 32 + (size_t)k * 32 + (n & 31)]
                  : W[(size_t)k * N + n];
  Wt[idx] = (bf16_t)v;
}

// ---------------------------------------------------------------------------
// LayerNorm: one wave per token row (C=512, 16 elems/lane), full-wave
// __shfl_xor reductions (warpSize==32 on gfx1250). Output bf16.
// ---------------------------------------------------------------------------
__global__ __launch_bounds__(256) void ln_kernel(const float* __restrict__ x,
                                                 const float* __restrict__ g,
                                                 const float* __restrict__ b,
                                                 bf16_t* __restrict__ out) {
  const int lane = threadIdx.x & 31;
  const int wid = threadIdx.x >> 5;
  const size_t row = (size_t)blockIdx.x * 8 + wid;
  const float* xr = x + row * Cn;
  const int c0 = lane * 16;

  float v[16];
#pragma unroll
  for (int i = 0; i < 4; ++i) {
    float4 t = *(const float4*)&xr[c0 + i * 4];
    v[i * 4 + 0] = t.x; v[i * 4 + 1] = t.y;
    v[i * 4 + 2] = t.z; v[i * 4 + 3] = t.w;
  }
  float s = 0.f;
#pragma unroll
  for (int i = 0; i < 16; ++i) s += v[i];
  s += __shfl_xor(s, 1); s += __shfl_xor(s, 2); s += __shfl_xor(s, 4);
  s += __shfl_xor(s, 8); s += __shfl_xor(s, 16);
  const float mu = s * (1.0f / Cn);

  float var = 0.f;
#pragma unroll
  for (int i = 0; i < 16; ++i) { float d = v[i] - mu; var += d * d; }
  var += __shfl_xor(var, 1); var += __shfl_xor(var, 2); var += __shfl_xor(var, 4);
  var += __shfl_xor(var, 8); var += __shfl_xor(var, 16);
  const float rstd = rsqrtf(var * (1.0f / Cn) + 1e-5f);

  v8bf o0, o1;
#pragma unroll
  for (int i = 0; i < 8; ++i)
    o0[i] = (bf16_t)((v[i] - mu) * rstd * g[c0 + i] + b[c0 + i]);
#pragma unroll
  for (int i = 0; i < 8; ++i)
    o1[i] = (bf16_t)((v[8 + i] - mu) * rstd * g[c0 + 8 + i] + b[c0 + 8 + i]);
  *(v8bf*)&out[row * Cn + c0] = o0;
  *(v8bf*)&out[row * Cn + c0 + 8] = o1;
}

// ---------------------------------------------------------------------------
// bf16 GEMM, fp32 accumulate: C(M,N) = A(M,K) @ Bt(N,K)^T [+bias][ReLU]
// [+resid]. Block tile 64x128, 4 waves (2 in M x 2 in N), each wave 32x64:
// 2 A-frags x 4 B-frags -> 8 WMMAs per 32-wide K-step. LDS double-buffered,
// filled with async global->LDS b128 copies; one barrier per K-step.
// ---------------------------------------------------------------------------
template <bool RELU, bool RESID, bool BIAS, typename OUT>
__global__ __launch_bounds__(128) void gemm_bf16(const bf16_t* __restrict__ A,
                                                 const bf16_t* __restrict__ Bt,
                                                 const float* __restrict__ bias,
                                                 const float* __restrict__ resid,
                                                 OUT* __restrict__ C,
                                                 int N, int K) {
  __shared__ __align__(16) bf16_t sA[2][64 * 32];    // 8 KB
  __shared__ __align__(16) bf16_t sB[2][128 * 32];   // 16 KB
  const int tid = threadIdx.x;
  const int lane = tid & 31;
  const int wid = tid >> 5;
  const int m0 = blockIdx.y * 64;
  const int n0 = blockIdx.x * 128;
  const int mW = (wid >> 1) * 32;
  const int nW = (wid & 1) * 64;

  // Issue the async tile loads for one K-step: 6 x b128 per thread
  // (A: 64x32 halves = 256 chunks, B: 128x32 halves = 512 chunks).
  auto stage = [&](int buf, int k0) {
#pragma unroll
    for (int i = 0; i < 2; ++i) {
      int c = tid + i * 128;
      int row = c >> 2, off = (c & 3) * 8;
      async_cp_b128(&A[(size_t)(m0 + row) * K + k0 + off],
                    &sA[buf][row * 32 + off]);
    }
#pragma unroll
    for (int i = 0; i < 4; ++i) {
      int c = tid + i * 128;
      int row = c >> 2, off = (c & 3) * 8;
      async_cp_b128(&Bt[(size_t)(n0 + row) * K + k0 + off],
                    &sB[buf][row * 32 + off]);
    }
  };

  v8f acc[2][4] = {};

  stage(0, 0);
  wait_async();
  __syncthreads();

  int buf = 0;
  for (int k0 = 0; k0 < K; k0 += 32) {
    if (k0 + 32 < K) stage(buf ^ 1, k0 + 32);  // overlap with compute

    v16bf a0 = ldfrag(sA[buf], mW, lane);
    v16bf a1 = ldfrag(sA[buf], mW + 16, lane);
#pragma unroll
    for (int tN = 0; tN < 4; ++tN) {
      v16bf bN = ldfrag(sB[buf], nW + tN * 16, lane);
      acc[0][tN] = WMMA_BF16(a0, bN, acc[0][tN]);
      acc[1][tN] = WMMA_BF16(a1, bN, acc[1][tN]);
    }

    if (k0 + 32 < K) wait_async();
    __syncthreads();
    buf ^= 1;
  }

  // Epilogue. C/D layout: lane l, VGPR r -> (m = r + 8*(l>>4), n = l&15).
  const int rhalf = 8 * (lane >> 4);
  const int ncol = lane & 15;
#pragma unroll
  for (int tM = 0; tM < 2; ++tM) {
#pragma unroll
    for (int tN = 0; tN < 4; ++tN) {
#pragma unroll
      for (int r = 0; r < 8; ++r) {
        int m = m0 + mW + tM * 16 + r + rhalf;
        int n = n0 + nW + tN * 16 + ncol;
        float v = acc[tM][tN][r];
        if (BIAS) v += bias[n];
        if (RELU) v = fmaxf(v, 0.f);
        size_t idx = (size_t)m * N + n;
        if (RESID) v += resid[idx];
        C[idx] = (OUT)v;
      }
    }
  }
}

// ---------------------------------------------------------------------------
// Flash attention, causal. Q/K/V/O stored (B, T, H*D) bf16, per-head base
// offset h*32, row stride 512. One block per (b, h, 64-query tile); 4 waves,
// 16 query rows each. D=32 == WMMA K, so QK^T is one WMMA per 16x16 tile.
// ---------------------------------------------------------------------------
__global__ __launch_bounds__(128) void attn_kernel(const bf16_t* __restrict__ Q,
                                                   const bf16_t* __restrict__ Km,
                                                   const bf16_t* __restrict__ V,
                                                   bf16_t* __restrict__ O) {
  __shared__ __align__(16) bf16_t sVT[32 * 32];      // V^T tile: [d][key]
  __shared__ __align__(16) bf16_t sP[4][16 * 32];    // per-wave P tile

  const int tid = threadIdx.x;
  const int lane = tid & 31;
  const int wid = tid >> 5;
  const int b = blockIdx.z;
  const int h = blockIdx.y;
  const int qblk = blockIdx.x * 64;
  const size_t base = ((size_t)b * Tn) * Cn + (size_t)h * Dn;
  const int qr0 = qblk + wid * 16;
  const int rhalf = 8 * (lane >> 4);
  const int ncol = lane & 15;

  // Q fragment (16 rows x 32 dims), loaded once, straight from global.
  v16bf aQ = ldfrag_g(Q + base, Cn, qr0, lane);

  v8f O0 = {}, O1 = {};
  float mrow[8], lrow[8];
#pragma unroll
  for (int r = 0; r < 8; ++r) { mrow[r] = -1e30f; lrow[r] = 0.f; }

  const float scale = 0.17677669529663687f;  // 1/sqrt(32)
  const float L2E = 1.4426950408889634f;

  for (int s0 = 0; s0 < qblk + 64; s0 += 32) {
    __syncthreads();
    {  // Stage V^T (32 keys x 32 dims, transposed) into LDS.
      int key = tid >> 2, off = (tid & 3) * 8;
      v8bf vv = *(const v8bf*)&V[base + (size_t)(s0 + key) * Cn + off];
#pragma unroll
      for (int j = 0; j < 8; ++j) sVT[(off + j) * 32 + key] = vv[j];
    }
    __syncthreads();

    // Scores S = Q @ K^T for 32 keys (K rows are already (N,K)-friendly).
    v16bf bK0 = ldfrag_g(Km + base, Cn, s0, lane);
    v16bf bK1 = ldfrag_g(Km + base, Cn, s0 + 16, lane);
    v8f S0 = {}, S1 = {};
    S0 = WMMA_BF16(aQ, bK0, S0);
    S1 = WMMA_BF16(aQ, bK1, S1);

    // Online softmax (row stats replicated over each 16-lane half).
#pragma unroll
    for (int r = 0; r < 8; ++r) {
      int qrow = qr0 + r + rhalf;
      float x0 = (s0 + ncol <= qrow) ? S0[r] * scale : -1e30f;
      float x1 = (s0 + 16 + ncol <= qrow) ? S1[r] * scale : -1e30f;
      float mx = fmaxf(x0, x1);
      mx = fmaxf(mx, __shfl_xor(mx, 1));
      mx = fmaxf(mx, __shfl_xor(mx, 2));
      mx = fmaxf(mx, __shfl_xor(mx, 4));
      mx = fmaxf(mx, __shfl_xor(mx, 8));
      float mnew = fmaxf(mrow[r], mx);
      float p0 = exp2f((x0 - mnew) * L2E);
      float p1 = exp2f((x1 - mnew) * L2E);
      float alpha = exp2f((mrow[r] - mnew) * L2E);
      float ps = p0 + p1;
      ps += __shfl_xor(ps, 1); ps += __shfl_xor(ps, 2);
      ps += __shfl_xor(ps, 4); ps += __shfl_xor(ps, 8);
      lrow[r] = lrow[r] * alpha + ps;
      mrow[r] = mnew;
      O0[r] *= alpha;
      O1[r] *= alpha;
      // Stash P in wave-private LDS to re-enter A-fragment layout.
      sP[wid][(r + rhalf) * 32 + ncol] = (bf16_t)p0;
      sP[wid][(r + rhalf) * 32 + 16 + ncol] = (bf16_t)p1;
    }

    // O += P @ V
    v16bf aP = ldfrag(sP[wid], 0, lane);
    v16bf bV0 = ldfrag(sVT, 0, lane);
    v16bf bV1 = ldfrag(sVT, 16, lane);
    O0 = WMMA_BF16(aP, bV0, O0);
    O1 = WMMA_BF16(aP, bV1, O1);
  }

#pragma unroll
  for (int r = 0; r < 8; ++r) {
    float inv = 1.0f / lrow[r];
    int qrow = qr0 + r + rhalf;
    bf16_t* op = O + base + (size_t)qrow * Cn;
    op[ncol] = (bf16_t)(O0[r] * inv);
    op[16 + ncol] = (bf16_t)(O1[r] * inv);
  }
}

// ---------------------------------------------------------------------------
// Host launcher
// ---------------------------------------------------------------------------
extern "C" void kernel_launch(void* const* d_in, const int* in_sizes, int n_in,
                              void* d_out, int out_size, void* d_ws, size_t ws_size,
                              hipStream_t stream) {
  (void)in_sizes; (void)n_in; (void)out_size; (void)ws_size;

  const float* x   = (const float*)d_in[0];
  const float* Wq  = (const float*)d_in[1];
  const float* Wk  = (const float*)d_in[2];
  const float* Wv  = (const float*)d_in[3];
  const float* Wp  = (const float*)d_in[4];
  const float* bp  = (const float*)d_in[5];
  const float* W1  = (const float*)d_in[6];
  const float* b1  = (const float*)d_in[7];
  const float* W2  = (const float*)d_in[8];
  const float* b2  = (const float*)d_in[9];
  const float* g1  = (const float*)d_in[10];
  const float* bl1 = (const float*)d_in[11];
  const float* g2  = (const float*)d_in[12];
  const float* bl2 = (const float*)d_in[13];
  float* out = (float*)d_out;
  char* ws = (char*)d_ws;

  const size_t MB = 1ull << 20;
  bf16_t* h_bf = (bf16_t*)(ws);             // 32 MB (LN1 then LN2 output)
  bf16_t* qb   = (bf16_t*)(ws + 32 * MB);   // 32 MB
  bf16_t* kb   = (bf16_t*)(ws + 64 * MB);   // 32 MB
  bf16_t* vb   = (bf16_t*)(ws + 96 * MB);   // 32 MB
  bf16_t* ao   = (bf16_t*)(ws + 128 * MB);  // 32 MB
  bf16_t* act  = qb;                        // 128 MB, reuses dead q/k/v/ao
  bf16_t* wq_t = (bf16_t*)(ws + 160 * MB);
  bf16_t* wk_t = wq_t + 512 * 512;
  bf16_t* wv_t = wk_t + 512 * 512;
  bf16_t* wp_t = wv_t + 512 * 512;
  bf16_t* w1_t = wp_t + 512 * 512;          // (2048, 512)
  bf16_t* w2_t = w1_t + 512 * 2048;         // (512, 2048)

  // 1) Weights -> bf16, transposed to (N, K).
  cvtw_kernel<true ><<<(512 * 512) / 256, 256, 0, stream>>>(Wq, wq_t, 512, 512);
  cvtw_kernel<true ><<<(512 * 512) / 256, 256, 0, stream>>>(Wk, wk_t, 512, 512);
  cvtw_kernel<true ><<<(512 * 512) / 256, 256, 0, stream>>>(Wv, wv_t, 512, 512);
  cvtw_kernel<false><<<(512 * 512) / 256, 256, 0, stream>>>(Wp, wp_t, 512, 512);
  cvtw_kernel<false><<<(512 * 2048) / 256, 256, 0, stream>>>(W1, w1_t, 512, 2048);
  cvtw_kernel<false><<<(2048 * 512) / 256, 256, 0, stream>>>(W2, w2_t, 2048, 512);

  // 2) LN1
  ln_kernel<<<Mrows / 8, 256, 0, stream>>>(x, g1, bl1, h_bf);

  // 3) QKV projections (stored (B, T, H, D) so attention sees stride-C heads)
  dim3 gq(512 / 128, Mrows / 64);
  gemm_bf16<false, false, false, bf16_t><<<gq, 128, 0, stream>>>(h_bf, wq_t, nullptr, nullptr, qb, 512, 512);
  gemm_bf16<false, false, false, bf16_t><<<gq, 128, 0, stream>>>(h_bf, wk_t, nullptr, nullptr, kb, 512, 512);
  gemm_bf16<false, false, false, bf16_t><<<gq, 128, 0, stream>>>(h_bf, wv_t, nullptr, nullptr, vb, 512, 512);

  // 4) Causal flash attention
  dim3 ga(Tn / 64, Hn, Bn);
  attn_kernel<<<ga, 128, 0, stream>>>(qb, kb, vb, ao);

  // 5) Output projection + bias + residual(x) -> d_out (fp32 residual stream)
  gemm_bf16<false, true, true, float><<<gq, 128, 0, stream>>>(ao, wp_t, bp, x, out, 512, 512);

  // 6) LN2 on residual stream
  ln_kernel<<<Mrows / 8, 256, 0, stream>>>(out, g2, bl2, h_bf);

  // 7) FFN up: ReLU(h2 @ W1 + b1) -> bf16
  dim3 gf(2048 / 128, Mrows / 64);
  gemm_bf16<true, false, true, bf16_t><<<gf, 128, 0, stream>>>(h_bf, w1_t, b1, nullptr, act, 2048, 512);

  // 8) FFN down + bias + residual(d_out) -> d_out
  gemm_bf16<false, true, true, float><<<gq, 128, 0, stream>>>(act, w2_t, b2, out, out, 512, 2048);
}
